// Linear_2834678415798
// MI455X (gfx1250) — 8-bit matrix operand order repaired and checked by reading, compiled, not run on hardware
//
#include <hip/hip_runtime.h>
#include <hip/hip_bf16.h>


typedef __attribute__((ext_vector_type(16))) int          v16i;
typedef __attribute__((ext_vector_type(8)))  float        v8f;
typedef __attribute__((ext_vector_type(4)))  unsigned int u32x4;
typedef __attribute__((ext_vector_type(8)))  int          i32x8;
typedef __attribute__((ext_vector_type(4)))  int          i32x4;

__device__ __forceinline__ unsigned int f32_to_e4m3_byte(float x) {
    unsigned int u = __float_as_uint(x);
    unsigned int s = (u >> 24) & 0x80u;
    unsigned int a = u & 0x7FFFFFFFu;
    if (a >= 0x7F800000u) return s | 0x7Fu;
    int e = (int)(a >> 23) - 127;
    if (e >= -6) {
        unsigned int lsb = (a >> 20) & 1u;
        unsigned int r = a + 0x0007FFFFu + lsb;
        r >>= 20;
        int e2 = (int)(r >> 3) - 127;
        unsigned int m = r & 7u;
        if (e2 > 8 || (e2 == 8 && m == 7u)) return s | 0x7Eu;
        return s | ((unsigned int)(e2 + 7) << 3) | m;
    }
    float af = __uint_as_float(a);
    int q = (int)(af * 512.0f + 0.5f);
    if (q >= 8) return s | 0x08u;
    return s | (unsigned int)q;
}

__device__ __forceinline__ unsigned int pack4_e4m3(float4 v) {
    return  f32_to_e4m3_byte(v.x)
         | (f32_to_e4m3_byte(v.y) << 8)
         | (f32_to_e4m3_byte(v.z) << 16)
         | (f32_to_e4m3_byte(v.w) << 24);
}

__global__ __launch_bounds__(256) void quant_swz_a_kernel(
    const float4* __restrict__ x, unsigned int* __restrict__ q,
    int rows, int K) {
    const int K4 = K >> 2;
    int idx = blockIdx.x * 256 + threadIdx.x;
    if (idx >= rows * K4) return;
    int m = idx / K4;
    int k = (idx - m * K4) << 2;

    unsigned int packed = pack4_e4m3(x[idx]);

    const int ktiles = K >> 7;
    const int kp = k & 127;
    unsigned int tile = (unsigned int)(m >> 4) * (unsigned int)ktiles + (unsigned int)(k >> 7);
    int l  = (m & 15) | (((kp >> 3) & 1) << 4);
    int pr = ((kp >> 6) << 2) | ((kp >> 4) & 3);
    int b  = (pr << 3) | (kp & 7);
    size_t byteoff = (size_t)tile * 2048 + (size_t)l * 64 + (size_t)b;
    q[byteoff >> 2] = packed;
}

__global__ __launch_bounds__(256) void quant_swz_b_kernel(
    const float4* __restrict__ x, unsigned int* __restrict__ q,
    int rows, int K) {
    const int K4 = K >> 2;
    int idx = blockIdx.x * 256 + threadIdx.x;
    if (idx >= rows * K4) return;
    int n = idx / K4;
    int k = (idx - n * K4) << 2;

    unsigned int packed = pack4_e4m3(x[idx]);

    const int ktiles = K >> 7;
    const int kp = k & 127;
    unsigned int tile = (unsigned int)(n >> 4) * (unsigned int)ktiles + (unsigned int)(k >> 7);
    int l = (n & 15) | (((kp >> 3) & 1) << 4);
    int b = ((kp >> 6) << 5) | (((kp >> 4) & 3) << 3) | (kp & 7);
    size_t byteoff = (size_t)tile * 2048 + (size_t)l * 64 + (size_t)b;
    q[byteoff >> 2] = packed;
}

__device__ __forceinline__ void tdm_load_2d(unsigned int lds_off,
                                            const void* gptr,
                                            unsigned int rows,
                                            unsigned long long row_stride_bytes) {
    unsigned long long ga = (unsigned long long)(size_t)gptr;
    u32x4 g0;
    g0[0] = 1u;
    g0[1] = lds_off;
    g0[2] = (unsigned int)(ga & 0xFFFFFFFFu);
    g0[3] = (unsigned int)((ga >> 32) & 0x1FFFFFFu)
          | (2u << 30);

    const unsigned int tdim0 = 0x40000000u;
    const unsigned int tdim1 = 0x40000000u;
    const unsigned int tile0 = 2048u >> 3;
    const unsigned int tile1 = rows;
    unsigned long long s0 = row_stride_bytes >> 3;

    i32x8 g1;
    g1[0] = (int)(3u << 16);
    g1[1] = (int)((tdim0 & 0xFFFFu) << 16);
    g1[2] = (int)(((tdim0 >> 16) & 0xFFFFu) | ((tdim1 & 0xFFFFu) << 16));
    g1[3] = (int)(((tdim1 >> 16) & 0xFFFFu) | (tile0 << 16));
    g1[4] = (int)(tile1 & 0xFFFFu);
    g1[5] = (int)(unsigned int)(s0 & 0xFFFFFFFFu);
    g1[6] = (int)(unsigned int)((s0 >> 32) & 0xFFFFu);
    g1[7] = 0;

    i32x4 g2 = {0, 0, 0, 0};
    i32x4 g3 = {0, 0, 0, 0};
    i32x8 g4 = {0, 0, 0, 0, 0, 0, 0, 0};
    __builtin_amdgcn_tensor_load_to_lds(g0, g1, g2, g3, g4, 0);
}

__global__ __launch_bounds__(256) void fp8_gemm_tdm_kernel(
    const unsigned char* __restrict__ qa,
    const unsigned char* __restrict__ qb,
    const float*         __restrict__ bias,
    float*               __restrict__ out,
    int T, int N, int K) {

    extern __shared__ unsigned char smem[];
    const unsigned int STAGE = 49152u;

    const int lane = threadIdx.x & 31;
    const int wave = threadIdx.x >> 5;
    const int wrow = wave >> 2;
    const int wcol = wave & 3;

    const int row0 = blockIdx.y * 128 + wrow * 64;
    const int col0 = blockIdx.x * 256 + wcol * 64;
    const int ml = lane & 15;
    const int ch = lane >> 4;

    const int    ktiles  = K >> 7;
    const size_t tstride = (size_t)ktiles * 2048;

    const unsigned char* agbase = qa + (size_t)(blockIdx.y * 8)  * tstride;
    const unsigned char* bgbase = qb + (size_t)(blockIdx.x * 16) * tstride;
    const unsigned int   sbase  = (unsigned int)(size_t)smem;

    v8f acc[4][4] = {};

    if (wave == 0) {
        tdm_load_2d(sbase,          agbase, 8,  tstride);
        tdm_load_2d(sbase + 16384u, bgbase, 16, tstride);
        __builtin_amdgcn_s_wait_tensorcnt(0);
    }
    __syncthreads();

    for (int kt = 0; kt < ktiles; ++kt) {
        const int cur = kt & 1;
        if (wave == 0 && (kt + 1) < ktiles) {
            const size_t go = (size_t)(kt + 1) * 2048;
            const unsigned int sb = sbase + (unsigned int)(cur ^ 1) * STAGE;
            tdm_load_2d(sb,          agbase + go, 8,  tstride);
            tdm_load_2d(sb + 16384u, bgbase + go, 16, tstride);
        }

        const unsigned char* As = smem + (size_t)cur * STAGE
                                + (size_t)(wrow * 4) * 2048 + (size_t)lane * 64;
        const unsigned char* Bs = smem + (size_t)cur * STAGE + 16384
                                + (size_t)(wcol * 4) * 2048 + (size_t)lane * 64;
        v16i a[4];
#pragma unroll
        for (int i = 0; i < 4; ++i)
            a[i] = *(const v16i*)(As + i * 2048);
#pragma unroll
        for (int j = 0; j < 4; ++j) {
            v16i b = *(const v16i*)(Bs + j * 2048);
#pragma unroll
            for (int i = 0; i < 4; ++i)
                acc[i][j] = __builtin_amdgcn_wmma_f32_16x16x128_fp8_fp8(
                    a[i], b, (short)0, acc[i][j], false, false);
        }

        if (wave == 0) __builtin_amdgcn_s_wait_tensorcnt(0);
        __syncthreads();
    }

#pragma unroll
    for (int j = 0; j < 4; ++j) {
        const int col = col0 + j * 16 + ml;
        const float bj = bias[col];
#pragma unroll
        for (int i = 0; i < 4; ++i) {
            const int rbase = row0 + i * 16 + ch * 8;
            float* op = out + (size_t)rbase * N + col;
#pragma unroll
            for (int r = 0; r < 8; ++r)
                op[(size_t)r * N] = acc[i][j][r] + bj;
        }
    }
}

extern "C" void kernel_launch(void* const* d_in, const int* in_sizes, int n_in,
                              void* d_out, int out_size, void* d_ws, size_t ws_size,
                              hipStream_t stream) {
    const float* inp    = (const float*)d_in[0];
    const float* weight = (const float*)d_in[1];
    const float* bias   = (const float*)d_in[2];
    float*       out    = (float*)d_out;

    const int N = in_sizes[2];
    const int K = in_sizes[1] / N;
    const int T = in_sizes[0] / K;

    unsigned char* qa = (unsigned char*)d_ws;
    unsigned char* qb = qa + (size_t)T * (size_t)K;

    {
        const int n4a = (T * K) / 4;
        const int n4b = (N * K) / 4;
        quant_swz_a_kernel<<<(n4a + 255) / 256, 256, 0, stream>>>(
            (const float4*)inp, (unsigned int*)qa, T, K);
        quant_swz_b_kernel<<<(n4b + 255) / 256, 256, 0, stream>>>(
            (const float4*)weight, (unsigned int*)qb, N, K);
    }

    {
        const int smem_bytes = 2 * 49152;
        (void)hipFuncSetAttribute((const void*)fp8_gemm_tdm_kernel,
                                  hipFuncAttributeMaxDynamicSharedMemorySize,
                                  smem_bytes);
        dim3 grid(N / 256, T / 128);
        dim3 block(256);
        fp8_gemm_tdm_kernel<<<grid, block, smem_bytes, stream>>>(
            qa, qb, bias, out, T, N, K);
    }
}
